// STGTCN_layer_60601988547101
// MI455X (gfx1250) — matrix operand order repaired and checked by reading, not compiled, not run on hardware
//
#include <hip/hip_runtime.h>
#include <hip/hip_bf16.h>
#include <math.h>

// ---------------------------------------------------------------------------
// ST-GCN layer for MI455X (gfx1250, wave32, WMMA + TDM).
//   x:(32,32,500,96) f32 -> out:(32,64,500,94) f32
// All three GEMM-shaped stages on the matrix pipe:
//   K1 temporal conv  : 3 shifted K=32 bf16 WMMA GEMMs + fused gate epilogue
//   K2/K3 diffusion   : batched 512x96x512 bf16 WMMA GEMM (TDM-staged LDS)
//   K4 projection     : 64x96 x (96 x v) bf16 WMMA GEMM + bias/PReLU epilogue
// Intermediate layout (bf16): buf[b][l][v], b=n*32+c (1024), l in [0,96), v in [0,512)
//   pad cells (l>=94, v>=500) are exact zeros -> padded WMMA tiles are exact.
// Workspace: ATpad 512KB + 3 x 96MB  (~303 MB total)
// ---------------------------------------------------------------------------

typedef __attribute__((ext_vector_type(16))) __bf16 v16bf;
typedef __attribute__((ext_vector_type(8)))  float  v8f;

union ABfrag { uint4 u[2]; v16bf v; };

__device__ __forceinline__ unsigned short f2bf(float f) {
  unsigned u = __float_as_uint(f);
  unsigned r = (u + 0x7FFFu + ((u >> 16) & 1u)) >> 16;   // round-to-nearest-even
  return (unsigned short)r;
}
__device__ __forceinline__ float bf2f(unsigned short h) {
  return __uint_as_float(((unsigned)h) << 16);
}
// fast reciprocal: single v_rcp_f32 (output is rounded to bf16 anyway)
__device__ __forceinline__ float frcp(float f) {
#if __has_builtin(__builtin_amdgcn_rcpf)
  return __builtin_amdgcn_rcpf(f);
#else
  return 1.f / f;
#endif
}
// zero the halfwords of a packed bf16 pair whose v-index >= nvalid
__device__ __forceinline__ unsigned maskp(unsigned w, int lo, int nv) {
  if (nv >= lo + 2) return w;
  if (nv == lo + 1) return w & 0xffffu;
  return 0u;
}

#define NB   32
#define CIN  32
#define VV   500
#define TT   96
#define TPV  94
#define COUT 64
#define VP   512           // padded V
#define LP   96            // padded L (= TT)
#define BATCH (NB*CIN)     // 1024
#define ROWP 520           // k_gemm LDS row pitch in bf16 (bank-conflict free)
#define XTP  40            // k_tconv xT row pitch (32 ci + 8 pad)
#define HTP  104           // k_project hlT row pitch (96 c + 8 pad)

// ---------------------------------------------------------------------------
// Tensor Data Mover staging (6-arg builtin on this toolchain), fallback = copy.
// ---------------------------------------------------------------------------
#if defined(__gfx1250__) && __has_builtin(__builtin_amdgcn_tensor_load_to_lds)
#define HAVE_TDM 1
typedef __attribute__((ext_vector_type(4))) unsigned int u32x4;
typedef __attribute__((ext_vector_type(8))) int i32x8;
typedef __attribute__((ext_vector_type(4))) int i32x4;

// 2D tile load: tile0 elems/row (2B each), tile1 rows, row stride stride0
// (in elements), optional LDS pad control bits in w0extra (pad_enable etc.)
__device__ __forceinline__ void tdm_load_2d(void* lds, const void* gsrc,
                                            unsigned tile0, unsigned tile1,
                                            unsigned long long stride0,
                                            unsigned w0extra) {
  unsigned lo = (unsigned)(uintptr_t)lds;            // low 32b of LDS aperture
  unsigned long long ga = (unsigned long long)(uintptr_t)gsrc;
  u32x4 g0 = { 1u,                                    // count=1, user descriptor
               lo,                                    // lds_addr [63:32]
               (unsigned)ga,                          // global_addr [95:64]
               (unsigned)(((ga >> 32) & 0x01FFFFFFull) | (2u << 30)) }; // +type=2
  i32x8 g1;
  g1[0] = (int)((1u << 16) | w0extra);               // data_size=2B, pad ctl
  g1[1] = (int)(tile0 << 16);                        // tensor_dim0[15:0] @ [63:48]
  g1[2] = (int)((tile0 >> 16) | (tile1 << 16));      // t_dim0 hi, tensor_dim1 lo
  g1[3] = (int)((tile1 >> 16) | (tile0 << 16));      // t_dim1 hi, tile_dim0
  g1[4] = (int)(tile1 & 0xFFFFu);                    // tile_dim1 (tile_dim2=0)
  g1[5] = (int)(unsigned)(stride0 & 0xFFFFFFFFull);  // tensor_dim0_stride lo32
  g1[6] = (int)(unsigned)(stride0 >> 32);            // stride hi16 (dim1_stride=0)
  g1[7] = 0;
  i32x4 z4 = { 0, 0, 0, 0 };
  i32x8 z8 = { 0, 0, 0, 0, 0, 0, 0, 0 };
  __builtin_amdgcn_tensor_load_to_lds(g0, g1, z4, z4, z8, 0);
}
#else
#define HAVE_TDM 0
#endif

// ---------------------------------------------------------------------------
// K0: ATpad[w][v] = A[v][w], padded to 512x512, bf16
// ---------------------------------------------------------------------------
__global__ void k_prep_A(const float* __restrict__ A, unsigned short* __restrict__ ATp) {
  int i = blockIdx.x * 256 + threadIdx.x;        // 0 .. 512*512-1
  int w = i >> 9, v = i & 511;
  unsigned short val = 0;
  if (v < VV && w < VV) val = f2bf(A[v * VV + w]);
  ATp[i] = val;
}

// ---------------------------------------------------------------------------
// K1: temporal gated conv as 3 shifted WMMA GEMMs -> xt[b][l][v] bf16
//   block = (n, v-chunk of 8), 256 threads (8 waves), grid = 32*64
//   per v: D[co][l] = sum_k W[:,:,k](32x32) x x[:, l+k]   (K = 32 = one step)
//   launch_bounds(256,1): keep 9 A-fragments + accs resident (no spills);
//   occupancy is LDS-limited (~135 KB) anyway.
// ---------------------------------------------------------------------------
__global__ void __launch_bounds__(256, 1)
k_tconv(const float* __restrict__ x,
        const float* __restrict__ w1, const float* __restrict__ b1,
        const float* __restrict__ w2, const float* __restrict__ b2,
        const float* __restrict__ w3, const float* __restrict__ b3,
        unsigned short* __restrict__ xtbuf) {
  extern __shared__ char smem[];
  unsigned short* xT = (unsigned short*)smem;          // [vi=8][t=98][ci pitch 40]
  unsigned short* wT = xT + 8 * 98 * XTP;              // [i=3][k=3][co=32][ci pitch 40]
  float*          bs = (float*)(wT + 9 * 32 * XTP);    // 96 f32
  unsigned short* yt = (unsigned short*)(bs + 96);     // [co=32][l=96][vi=8]
  const int tid = threadIdx.x;
  const int n = blockIdx.x >> 6, vc = blockIdx.x & 63, v0 = vc * 8;

  // ---- stage weights (bf16, [i][k][co][ci]) + biases ----
  for (int e = tid; e < 3072; e += 256) {
    int co = e / 96, r = e % 96, ci = r / 3, k = r % 3;
    wT[((0 * 3 + k) * 32 + co) * XTP + ci] = f2bf(w1[e]);
    wT[((1 * 3 + k) * 32 + co) * XTP + ci] = f2bf(w2[e]);
    wT[((2 * 3 + k) * 32 + co) * XTP + ci] = f2bf(w3[e]);
  }
  if (tid < 32) { bs[tid] = b1[tid]; bs[32 + tid] = b2[tid]; bs[64 + tid] = b3[tid]; }
  // ---- stage x transposed: xT[vi][t][ci], t rows 96..97 zeroed (shift pad) ----
  for (int i = tid; i < 32 * 8 * 96; i += 256) {
    int ci = i / 768, r = i % 768, vi = r / 96, t = r % 96;
    int v = v0 + vi;
    float xv = (v < VV) ? x[((size_t)(n * 32 + ci) * VV + v) * TT + t] : 0.f;
    xT[(vi * 98 + t) * XTP + ci] = f2bf(xv);
  }
  for (int i = tid; i < 8 * 2 * 32; i += 256) {
    int vi = i >> 6, tz = 96 + ((i >> 5) & 1), ci = i & 31;
    xT[(vi * 98 + tz) * XTP + ci] = 0;
  }
  __syncthreads();

  const int wave = tid >> 5, lane = tid & 31;
  const int ln = lane & 15, half = lane >> 4;

  for (int j = 0; j < 2; ++j) {
    const int job = wave * 2 + j;                 // 16 jobs: (vi, mt)
    const int vi = job >> 1, mt = job & 1;
    // 9 A fragments: W[i][k], row co = mt*16+ln, K(ci) split per A layout
    ABfrag afr[3][3];
#pragma unroll
    for (int i = 0; i < 3; ++i)
#pragma unroll
      for (int k = 0; k < 3; ++k) {
        const unsigned short* ab =
            wT + ((i * 3 + k) * 32 + mt * 16 + ln) * XTP + half * 8;
        afr[i][k].u[0] = ((const uint4*)ab)[0];        // ci = half*8 + [0..7]
        afr[i][k].u[1] = ((const uint4*)(ab + 16))[0]; // ci = 16+half*8 + [0..7]
      }
#pragma unroll 1
    for (int nt = 0; nt < 6; ++nt) {
      v8f a1, a2, a3;
#pragma unroll
      for (int r = 0; r < 8; ++r) { a1[r] = 0.f; a2[r] = 0.f; a3[r] = 0.f; }
#pragma unroll
      for (int k = 0; k < 3; ++k) {
        // B fragment: lane col l = nt*16+ln, K(ci) = half*8 + [0..7], 16+half*8 + [0..7]
        ABfrag bb;
        const uint4* bp = (const uint4*)(xT + (vi * 98 + nt * 16 + ln + k) * XTP + half * 8);
        bb.u[0] = bp[0]; bb.u[1] = bp[2];
        a1 = __builtin_amdgcn_wmma_f32_16x16x32_bf16(false, afr[0][k].v, false, bb.v,
                                                     (short)0, a1, false, false);
        a2 = __builtin_amdgcn_wmma_f32_16x16x32_bf16(false, afr[1][k].v, false, bb.v,
                                                     (short)0, a2, false, false);
        a3 = __builtin_amdgcn_wmma_f32_16x16x32_bf16(false, afr[2][k].v, false, bb.v,
                                                     (short)0, a3, false, false);
      }
      // epilogue: bias + gated combine + relu -> yt[co][l][vi]
      const int l = nt * 16 + ln;
#pragma unroll
      for (int r = 0; r < 8; ++r) {
        int co = mt * 16 + half * 8 + r;
        float c1 = a1[r] + bs[co];
        float c2 = a2[r] + bs[32 + co];
        float c3 = a3[r] + bs[64 + co];
        float sg = frcp(1.f + __expf(-c2));
        float val = fmaxf(c1 + sg + c3, 0.f);
        yt[(co * 96 + l) * 8 + vi] = f2bf(val);
      }
    }
  }
  __syncthreads();

  // ---- packed coalesced writeout with exact zero padding ----
  int nvalid = VV - v0; nvalid = nvalid < 0 ? 0 : (nvalid > 8 ? 8 : nvalid);
  for (int oi = tid; oi < 32 * 96; oi += 256) {
    int co = oi / 96, l = oi % 96;
    uint4 q = { 0, 0, 0, 0 };
    if (l < TPV) {
      q = *((const uint4*)(yt + (co * 96 + l) * 8));
      q.x = maskp(q.x, 0, nvalid);
      q.y = maskp(q.y, 2, nvalid);
      q.z = maskp(q.z, 4, nvalid);
      q.w = maskp(q.w, 6, nvalid);
    }
    *((uint4*)(xtbuf + (((size_t)(n * 32 + co) * LP + l) * VP + v0))) = q;
  }
}

// ---------------------------------------------------------------------------
// K2/K3: Y[b][l][w] = sum_v AT[w][v] * X[b][l][v]   (bf16 in, f32 acc, bf16 out)
//   block = batch b (1024 blocks), 256 threads = 8 waves
//   M=512 (w, 32 tiles, 2 per wave register-blocked), N=96 (l, 6 tiles),
//   K=512 (v, 16 steps of 32); LDS slab staged by TDM (pad-pitched rows)
// ---------------------------------------------------------------------------
__global__ void k_gemm_diffuse(const unsigned short* __restrict__ ATp,
                               const unsigned short* __restrict__ Xg,
                               unsigned short* __restrict__ Yg) {
  extern __shared__ char smem[];
  unsigned short* xl = (unsigned short*)smem;      // 96 rows x ROWP bf16
  const int tid = threadIdx.x, b = blockIdx.x;

#if HAVE_TDM
  if (tid < 32) {   // one wave issues the DMA; EXEC-independent, wave-ordered
    tdm_load_2d(xl, Xg + (size_t)b * (LP * VP), VP, LP, (unsigned long long)VP,
                (1u << 20) | (7u << 22) | (3u << 25)); // pad_en, 256 dw, +4 dw
    __builtin_amdgcn_s_wait_tensorcnt(0);
  }
#else
  {
    const uint4* src = (const uint4*)(Xg + (size_t)b * (LP * VP));
    for (int i = tid; i < 6144; i += 256) {
      int row = i >> 6, ch = i & 63;               // 64 x 16B chunks per row
      ((uint4*)(xl + row * ROWP))[ch] = src[i];
    }
  }
#endif
  __syncthreads();

  const int wave = tid >> 5, lane = tid & 31;
  const int ln = lane & 15, half = lane >> 4;

  for (int pair = 0; pair < 2; ++pair) {
    const int mt0 = wave + pair * 16;              // two M-tiles per wave
    const int mt1 = mt0 + 8;
    v8f accA[6], accB[6];
#pragma unroll
    for (int nt = 0; nt < 6; ++nt)
#pragma unroll
      for (int r = 0; r < 8; ++r) { accA[nt][r] = 0.f; accB[nt][r] = 0.f; }

    const unsigned short* arow0 = ATp + (size_t)(mt0 * 16 + ln) * VP + half * 8;
    const unsigned short* arow1 = arow0 + (size_t)(8 * 16) * VP;
    __builtin_prefetch(arow0, 0, 0);               // global_prefetch_b8 (L2-hot AT)
    __builtin_prefetch(arow1, 0, 0);

#pragma unroll 1
    for (int ks = 0; ks < 16; ++ks) {
      ABfrag a0, a1;
      const uint4* ap0 = (const uint4*)(arow0 + ks * 32);
      const uint4* ap1 = (const uint4*)(arow1 + ks * 32);
      a0.u[0] = ap0[0]; a0.u[1] = ap0[2];
      a1.u[0] = ap1[0]; a1.u[1] = ap1[2];
#pragma unroll
      for (int nt = 0; nt < 6; ++nt) {
        ABfrag bb;
        const uint4* bp = (const uint4*)(xl + (nt * 16 + ln) * ROWP + ks * 32 + half * 8);
        bb.u[0] = bp[0]; bb.u[1] = bp[2];
        accA[nt] = __builtin_amdgcn_wmma_f32_16x16x32_bf16(
            false, a0.v, false, bb.v, (short)0, accA[nt], false, false);
        accB[nt] = __builtin_amdgcn_wmma_f32_16x16x32_bf16(
            false, a1.v, false, bb.v, (short)0, accB[nt], false, false);
      }
    }
#pragma unroll
    for (int nt = 0; nt < 6; ++nt) {
      int lg = nt * 16 + ln;
      uint4 pk;
      pk.x = (unsigned)f2bf(accA[nt][0]) | ((unsigned)f2bf(accA[nt][1]) << 16);
      pk.y = (unsigned)f2bf(accA[nt][2]) | ((unsigned)f2bf(accA[nt][3]) << 16);
      pk.z = (unsigned)f2bf(accA[nt][4]) | ((unsigned)f2bf(accA[nt][5]) << 16);
      pk.w = (unsigned)f2bf(accA[nt][6]) | ((unsigned)f2bf(accA[nt][7]) << 16);
      *((uint4*)(Yg + ((size_t)b * LP + lg) * VP + mt0 * 16 + half * 8)) = pk;
      pk.x = (unsigned)f2bf(accB[nt][0]) | ((unsigned)f2bf(accB[nt][1]) << 16);
      pk.y = (unsigned)f2bf(accB[nt][2]) | ((unsigned)f2bf(accB[nt][3]) << 16);
      pk.z = (unsigned)f2bf(accB[nt][4]) | ((unsigned)f2bf(accB[nt][5]) << 16);
      pk.w = (unsigned)f2bf(accB[nt][6]) | ((unsigned)f2bf(accB[nt][7]) << 16);
      *((uint4*)(Yg + ((size_t)b * LP + lg) * VP + mt1 * 16 + half * 8)) = pk;
    }
  }
}

// ---------------------------------------------------------------------------
// K4: out[n][o][v][l] = prelu( b[o] + sum_c w_out[o][c] * h[c][v][l] )
//   h = [xt; x1; x2]; WMMA GEMM: M=64 (o, 4 tiles), K=96 (c, 3 steps),
//   N=512 (v, 32 tiles); block = (n,l), 256 threads, grid = 32*94
// ---------------------------------------------------------------------------
__global__ void k_project(const unsigned short* __restrict__ xt,
                          const unsigned short* __restrict__ x1,
                          const unsigned short* __restrict__ x2,
                          const float* __restrict__ w_out,
                          const float* __restrict__ b_out,
                          const float* __restrict__ prelu_a,
                          float* __restrict__ out) {
  extern __shared__ char smem[];
  unsigned short* hlT = (unsigned short*)smem;         // [v=512][c pitch 104]
  unsigned short* wT  = hlT + VP * HTP;                // [o=64][c pitch 104]
  float*          bo  = (float*)(wT + COUT * HTP);     // 64 f32
  const int tid = threadIdx.x;
  const int n = blockIdx.x / TPV, l = blockIdx.x % TPV;

  // ---- stage h transposed: hlT[v][c] from buf[b][l][v] rows ----
  for (int i = tid; i < 96 * 256; i += 256) {          // uint = 2 v per read
    int c = i >> 8, vp = i & 255;
    const unsigned short* srcb = (c < 32) ? xt : (c < 64) ? x1 : x2;
    const unsigned* row =
        (const unsigned*)(srcb + (((size_t)(n * 32 + (c & 31))) * LP + l) * VP);
    unsigned pv = row[vp];
    hlT[(2 * vp + 0) * HTP + c] = (unsigned short)(pv & 0xffffu);
    hlT[(2 * vp + 1) * HTP + c] = (unsigned short)(pv >> 16);
  }
  for (int i = tid; i < COUT * 96; i += 256) {
    int o = i / 96, c = i % 96;
    wT[o * HTP + c] = f2bf(w_out[i]);
  }
  if (tid < COUT) bo[tid] = b_out[tid];
  __syncthreads();

  const float pa = prelu_a[0];
  const int wave = tid >> 5, lane = tid & 31;
  const int ln = lane & 15, half = lane >> 4;

  for (int nt = wave; nt < 32; nt += 8) {              // v tiles
    // 3 B fragments for this v-column strip (shared across 4 M-tiles)
    ABfrag bfr[3];
#pragma unroll
    for (int ks = 0; ks < 3; ++ks) {
      const uint4* bp = (const uint4*)(hlT + (nt * 16 + ln) * HTP + ks * 32 + half * 8);
      bfr[ks].u[0] = bp[0]; bfr[ks].u[1] = bp[2];
    }
    const int v = nt * 16 + ln;
#pragma unroll 1
    for (int mt = 0; mt < 4; ++mt) {
      v8f acc;
#pragma unroll
      for (int r = 0; r < 8; ++r) acc[r] = 0.f;
#pragma unroll
      for (int ks = 0; ks < 3; ++ks) {
        ABfrag a;
        const unsigned short* ab = wT + (mt * 16 + ln) * HTP + ks * 32 + half * 8;
        a.u[0] = ((const uint4*)ab)[0];
        a.u[1] = ((const uint4*)(ab + 16))[0];
        acc = __builtin_amdgcn_wmma_f32_16x16x32_bf16(false, a.v, false, bfr[ks].v,
                                                      (short)0, acc, false, false);
      }
      if (v < VV) {                                    // gated epilogue stores
#pragma unroll
        for (int r = 0; r < 8; ++r) {
          int o = mt * 16 + half * 8 + r;
          float val = acc[r] + bo[o];
          val = val > 0.f ? val : pa * val;
          out[(((size_t)(n * COUT + o)) * VV + v) * TPV + l] = val;
        }
      }
    }
  }
}

// ---------------------------------------------------------------------------
extern "C" void kernel_launch(void* const* d_in, const int* in_sizes, int n_in,
                              void* d_out, int out_size, void* d_ws, size_t ws_size,
                              hipStream_t stream) {
  const float* x     = (const float*)d_in[0];
  const float* A     = (const float*)d_in[1];
  const float* w1    = (const float*)d_in[2];
  const float* b1    = (const float*)d_in[3];
  const float* w2    = (const float*)d_in[4];
  const float* b2    = (const float*)d_in[5];
  const float* w3    = (const float*)d_in[6];
  const float* b3    = (const float*)d_in[7];
  const float* w_out = (const float*)d_in[8];
  const float* b_out = (const float*)d_in[9];
  const float* pa    = (const float*)d_in[10];
  float* out = (float*)d_out;

  char* p = (char*)d_ws;
  unsigned short* ATp = (unsigned short*)p; p += (size_t)VP * VP * 2;
  const size_t bufBytes = (size_t)BATCH * LP * VP * 2;   // ~96 MB each
  unsigned short* xtb = (unsigned short*)p; p += bufBytes;
  unsigned short* x1b = (unsigned short*)p; p += bufBytes;
  unsigned short* x2b = (unsigned short*)p;

  k_prep_A<<<(VP * VP) / 256, 256, 0, stream>>>(A, ATp);

  // xT + wT + bias + yt
  size_t sh1 = (size_t)(8 * 98 * XTP) * 2 + (size_t)(9 * 32 * XTP) * 2
             + 96 * 4 + (size_t)(32 * 96 * 8) * 2;               // ~135 KB
  k_tconv<<<NB * 64, 256, sh1, stream>>>(x, w1, b1, w2, b2, w3, b3, xtb);

  size_t shg = (size_t)LP * ROWP * 2;                               // ~99.8 KB
  k_gemm_diffuse<<<BATCH, 256, shg, stream>>>(ATp, xtb, x1b);
  k_gemm_diffuse<<<BATCH, 256, shg, stream>>>(ATp, x1b, x2b);

  size_t shp = (size_t)VP * HTP * 2 + (size_t)COUT * HTP * 2 + COUT * 4; // ~120 KB
  k_project<<<NB * TPV, 256, shp, stream>>>(xtb, x1b, x2b, w_out, b_out, pa, out);
}
